// ReconstructionHead_20847771254987
// MI455X (gfx1250) — matrix operand order repaired and checked by reading, not compiled, not run on hardware
//
#include <hip/hip_runtime.h>
#include <math.h>

// ---------------- CDNA5 WMMA types ----------------
typedef __attribute__((ext_vector_type(16))) __bf16 v16bf;
typedef __attribute__((ext_vector_type(8)))  float  v8f;

#define EMBED 128
#define EQUI  16
#define INDIM 144
#define K1PAD 160          // 5 k-tiles of 32
#define HID   256          // 8 k-tiles of 32
#define SEQ   128
#define LN_EPS 1e-5f

#define A1_STRIDE 168      // ushort stride; 336B = 84 words, 84%64=20 -> conflict-free across 16 rows
#define A2_STRIDE 264      // 528B = 132 words, 132%64=4 -> conflict-free
#define H_STRIDE  260      // float stride; 260%64=4 -> conflict-free

#define W1_PACK_UINTS (5 * 16 * 32 * 8)   // 20480 uints (bf16 pairs)
#define W2_PACK_UINTS (8 * 16 * 32 * 8)   // 32768 uints

__device__ __forceinline__ unsigned short f2bf(float f) {
    unsigned int u = __float_as_uint(f);
    unsigned int r = u + 0x7FFFu + ((u >> 16) & 1u);   // round-to-nearest-even
    return (unsigned short)(r >> 16);
}

__device__ __forceinline__ float gelu_exact(float x) {
    return 0.5f * x * (1.0f + erff(x * 0.70710678118654752f));
}

struct __align__(16) U4 { unsigned int x, y, z, w; };

// ---- pack fp32 (K x 256) weights into WMMA B-fragment order, bf16 ----
// Layout: [ktile][ntile][lane 0..31][8 uints] ; lane holds col n = ntile*16 + (lane&15),
// rows k = ktile*32 + (lane>>4)*16 + 2*j  (bf16 pair per uint, lo=k, hi=k+1).
__global__ void pack_w_bf16_kernel(const float* __restrict__ W, unsigned int* __restrict__ dst,
                                   int Kdim, int total) {
    int g = blockIdx.x * blockDim.x + threadIdx.x;
    if (g >= total) return;
    int j    = g & 7;
    int lane = (g >> 3) & 31;
    int nt   = (g >> 8) & 15;
    int kt   = g >> 12;
    int n = nt * 16 + (lane & 15);
    int k = kt * 32 + (lane >> 4) * 16 + j * 2;
    float v0 = (k     < Kdim) ? W[(size_t)k       * HID + n] : 0.0f;
    float v1 = (k + 1 < Kdim) ? W[(size_t)(k + 1) * HID + n] : 0.0f;
    dst[g] = (unsigned int)f2bf(v0) | ((unsigned int)f2bf(v1) << 16);
}

// ---- A-fragment loader from LDS (16x32 A, contiguous k: half hh holds k = 16*hh..16*hh+15) ----
__device__ __forceinline__ v16bf load_afrag(const unsigned short* Ab, int stride,
                                            int m, int hh, int kbase) {
    union { unsigned int u[8]; v16bf v; } au;
    const unsigned short* row = Ab + m * stride + kbase;
#pragma unroll
    for (int j = 0; j < 8; ++j) {
        int lk = hh * 16 + j * 2;
        au.u[j] = *(const unsigned int*)(row + lk);
    }
    return au.v;
}

// ---- B-fragment loader from packed global (2 x b128 per lane) ----
__device__ __forceinline__ v16bf load_bfrag(const unsigned int* __restrict__ pW,
                                            int s, int nt, int lane) {
    union { U4 q[2]; v16bf v; } bu;
    const U4* p = (const U4*)(pW + ((((size_t)s * 16 + nt) * 32 + lane) << 3));
    bu.q[0] = p[0];
    bu.q[1] = p[1];
    return bu.v;
}

// ---- fused gather + MLP head ----
// block = 128 threads (4 waves, wave32), handles 16 batch rows; grid = B/16 = 1024
__global__ void __launch_bounds__(128)
recon_head_kernel(const float* __restrict__ x_embed,
                  const float* __restrict__ x_equi,
                  const int*   __restrict__ mask_idx,
                  const unsigned int* __restrict__ pW1,
                  const float* __restrict__ b1,
                  const float* __restrict__ ln_g,
                  const float* __restrict__ ln_b,
                  const unsigned int* __restrict__ pW2,
                  const float* __restrict__ b2,
                  const float* __restrict__ w3,
                  const float* __restrict__ b3,
                  float* __restrict__ out) {
    __shared__ unsigned short A1[16 * A1_STRIDE];   // gathered+padded input, bf16
    __shared__ unsigned short A2[16 * A2_STRIDE];   // post LN+GELU, bf16
    __shared__ float          hraw[16 * H_STRIDE];  // fp32 staging (h1 then h2)

    const int tid  = threadIdx.x;
    const int lane = tid & 31;
    const int wave = tid >> 5;
    const int m    = lane & 15;   // row within 16-row tile / col within C tile
    const int hh   = lane >> 4;   // lane half
    const int blk  = blockIdx.x;

    // ---- 1) gather 16 rows of concat(embed,equi) -> bf16 LDS, zero-pad to 160 ----
#pragma unroll
    for (int i = tid; i < 16 * K1PAD; i += 128) {
        int r = i / K1PAD;
        int c = i - r * K1PAD;
        int b = blk * 16 + r;
        int idx = mask_idx[b];
        float v;
        if (c < EMBED)      v = x_embed[((size_t)b * SEQ + idx) * EMBED + c];
        else if (c < INDIM) v = x_equi [((size_t)b * SEQ + idx) * EQUI  + (c - EMBED)];
        else                v = 0.0f;
        A1[r * A1_STRIDE + c] = f2bf(v);
    }
    __syncthreads();

    // ---- 2) GEMM1: h1(16x256) = A1(16x160) @ w1 ; wave covers N = wave*64..+64 ----
    v8f acc0 = {0,0,0,0,0,0,0,0}, acc1 = acc0, acc2 = acc0, acc3 = acc0;
#pragma unroll
    for (int s = 0; s < 5; ++s) {
        v16bf a = load_afrag(A1, A1_STRIDE, m, hh, s * 32);
        v16bf bb0 = load_bfrag(pW1, s, wave * 4 + 0, lane);
        acc0 = __builtin_amdgcn_wmma_f32_16x16x32_bf16(false, a, false, bb0, (short)0, acc0, false, false);
        v16bf bb1 = load_bfrag(pW1, s, wave * 4 + 1, lane);
        acc1 = __builtin_amdgcn_wmma_f32_16x16x32_bf16(false, a, false, bb1, (short)0, acc1, false, false);
        v16bf bb2 = load_bfrag(pW1, s, wave * 4 + 2, lane);
        acc2 = __builtin_amdgcn_wmma_f32_16x16x32_bf16(false, a, false, bb2, (short)0, acc2, false, false);
        v16bf bb3 = load_bfrag(pW1, s, wave * 4 + 3, lane);
        acc3 = __builtin_amdgcn_wmma_f32_16x16x32_bf16(false, a, false, bb3, (short)0, acc3, false, false);
    }
    // C layout (ISA 7.12.2): VGPR v -> row v + 8*hh ; col = lane&15 within tile
    {
        v8f* accs[4] = {&acc0, &acc1, &acc2, &acc3};
#pragma unroll
        for (int t = 0; t < 4; ++t) {
            int ncol = wave * 64 + t * 16 + m;
            float bias = b1[ncol];
#pragma unroll
            for (int v = 0; v < 8; ++v) {
                int row = v + 8 * hh;
                hraw[row * H_STRIDE + ncol] = (*accs[t])[v] + bias;
            }
        }
    }
    __syncthreads();

    // ---- 3) LayerNorm + exact GELU -> bf16 A2 ; wave handles rows wave, wave+4, +8, +12 ----
#pragma unroll
    for (int i = 0; i < 4; ++i) {
        int r = wave + 4 * i;
        float x[8];
        float sum = 0.0f, sq = 0.0f;
#pragma unroll
        for (int j = 0; j < 8; ++j) {
            x[j] = hraw[r * H_STRIDE + lane * 8 + j];
            sum += x[j];
            sq  += x[j] * x[j];
        }
#pragma unroll
        for (int off = 16; off > 0; off >>= 1) {
            sum += __shfl_xor(sum, off, 32);
            sq  += __shfl_xor(sq,  off, 32);
        }
        float mu   = sum * (1.0f / HID);
        float var  = sq * (1.0f / HID) - mu * mu;
        float rstd = rsqrtf(var + LN_EPS);
#pragma unroll
        for (int j = 0; j < 8; ++j) {
            int c = lane * 8 + j;
            float y = (x[j] - mu) * rstd * ln_g[c] + ln_b[c];
            A2[r * A2_STRIDE + c] = f2bf(gelu_exact(y));
        }
    }
    __syncthreads();

    // ---- 4) GEMM2: h2 = gelu(A2(16x256) @ w2 + b2) -> fp32 LDS ----
    acc0 = (v8f){0,0,0,0,0,0,0,0}; acc1 = acc0; acc2 = acc0; acc3 = acc0;
#pragma unroll
    for (int s = 0; s < 8; ++s) {
        v16bf a = load_afrag(A2, A2_STRIDE, m, hh, s * 32);
        v16bf bb0 = load_bfrag(pW2, s, wave * 4 + 0, lane);
        acc0 = __builtin_amdgcn_wmma_f32_16x16x32_bf16(false, a, false, bb0, (short)0, acc0, false, false);
        v16bf bb1 = load_bfrag(pW2, s, wave * 4 + 1, lane);
        acc1 = __builtin_amdgcn_wmma_f32_16x16x32_bf16(false, a, false, bb1, (short)0, acc1, false, false);
        v16bf bb2 = load_bfrag(pW2, s, wave * 4 + 2, lane);
        acc2 = __builtin_amdgcn_wmma_f32_16x16x32_bf16(false, a, false, bb2, (short)0, acc2, false, false);
        v16bf bb3 = load_bfrag(pW2, s, wave * 4 + 3, lane);
        acc3 = __builtin_amdgcn_wmma_f32_16x16x32_bf16(false, a, false, bb3, (short)0, acc3, false, false);
    }
    {
        v8f* accs[4] = {&acc0, &acc1, &acc2, &acc3};
#pragma unroll
        for (int t = 0; t < 4; ++t) {
            int ncol = wave * 64 + t * 16 + m;
            float bias = b2[ncol];
#pragma unroll
            for (int v = 0; v < 8; ++v) {
                int row = v + 8 * hh;
                hraw[row * H_STRIDE + ncol] = gelu_exact((*accs[t])[v] + bias);
            }
        }
    }
    __syncthreads();

    // ---- 5) final 256 -> 4 projection, fp32 VALU (1k MAC/row) ----
    if (tid < 64) {
        int r = tid >> 2;
        int o = tid & 3;
        float acc = b3[o];
#pragma unroll 8
        for (int k = 0; k < HID; ++k)
            acc += hraw[r * H_STRIDE + k] * w3[k * 4 + o];
        out[((size_t)blk * 16 + r) * 4 + o] = acc;
    }
}

extern "C" void kernel_launch(void* const* d_in, const int* in_sizes, int n_in,
                              void* d_out, int out_size, void* d_ws, size_t ws_size,
                              hipStream_t stream) {
    (void)in_sizes; (void)n_in; (void)out_size; (void)ws_size;
    const float* x_embed = (const float*)d_in[0];
    const float* x_equi  = (const float*)d_in[1];
    const int*   mask    = (const int*)d_in[2];
    const float* w1      = (const float*)d_in[3];
    const float* b1      = (const float*)d_in[4];
    const float* ln_g    = (const float*)d_in[5];
    const float* ln_b    = (const float*)d_in[6];
    const float* w2      = (const float*)d_in[7];
    const float* b2      = (const float*)d_in[8];
    const float* w3      = (const float*)d_in[9];
    const float* b3      = (const float*)d_in[10];
    float* out = (float*)d_out;

    unsigned int* pW1 = (unsigned int*)d_ws;           // 81920 B
    unsigned int* pW2 = pW1 + W1_PACK_UINTS;           // 131072 B (total ~208 KB of d_ws)

    // pack weights into WMMA-fragment-ordered bf16 (L2-resident thereafter)
    pack_w_bf16_kernel<<<(W1_PACK_UINTS + 255) / 256, 256, 0, stream>>>(w1, pW1, INDIM, W1_PACK_UINTS);
    pack_w_bf16_kernel<<<(W2_PACK_UINTS + 255) / 256, 256, 0, stream>>>(w2, pW2, HID,   W2_PACK_UINTS);

    // fused gather + MLP: 16384/16 = 1024 blocks x 128 threads (4 waves)
    recon_head_kernel<<<1024, 128, 0, stream>>>(x_embed, x_equi, mask,
                                                pW1, b1, ln_g, ln_b,
                                                pW2, b2, w3, b3, out);
}
